// SelfAttention_37658273251396
// MI455X (gfx1250) — matrix operand order repaired and checked by reading, not compiled, not run on hardware
//
#include <hip/hip_runtime.h>
#include <hip/hip_bf16.h>

typedef _Float16 v16h __attribute__((ext_vector_type(16)));
typedef _Float16 half8 __attribute__((ext_vector_type(8)));
typedef float    v8f  __attribute__((ext_vector_type(8)));

#define Bsz   2
#define Cdim  384
#define C3    1152
#define Sdim  65536
#define NHEAD 6
#define HD    64
#define CHK   128
#define NCHK  512   // Sdim / CHK

// ---------------- fragment helpers ----------------
static __device__ __forceinline__ v16h load_frag16(const _Float16* p0, const _Float16* p1) {
    union { v16h v; half8 h[2]; } u;
    u.h[0] = *(const half8*)p0;
    u.h[1] = *(const half8*)p1;
    return u.v;
}
static __device__ __forceinline__ v8f wmma16(v16h a, v16h b, v8f c) {
    return __builtin_amdgcn_wmma_f32_16x16x32_f16(false, a, false, b, (short)0, c, false, false);
}
static __device__ __forceinline__ v8f splat8(float x) {
    v8f r;
#pragma unroll
    for (int i = 0; i < 8; ++i) r[i] = x;
    return r;
}

// ---------------- kernel 0: weight transpose + f16 convert ----------------
__global__ void prep_weights(const float* __restrict__ wqkv, const float* __restrict__ wproj,
                             _Float16* __restrict__ wqkvT, _Float16* __restrict__ wprojT) {
    for (int i = blockIdx.x * blockDim.x + threadIdx.x; i < C3 * Cdim; i += gridDim.x * blockDim.x) {
        int col = i / Cdim, c = i % Cdim;
        wqkvT[i] = (_Float16)wqkv[c * C3 + col];          // wqkvT[col][c]
    }
    for (int i = blockIdx.x * blockDim.x + threadIdx.x; i < Cdim * Cdim; i += gridDim.x * blockDim.x) {
        int col = i / Cdim, c = i % Cdim;
        wprojT[i] = (_Float16)wproj[c * Cdim + col];      // wprojT[col][c]
    }
}

// ---------------- kernel 1: fused QKV GEMM + chunk attention ----------------
#define XS_STRIDE 392   // 384 + 8 pad (multiple of 8 halves -> 16B aligned frags)
#define QK_STRIDE 72    // 64 + 8
#define PV_STRIDE 136   // 128 + 8

// LDS layout (halves). Ps aliases Qs+Ks (dead after logits GEMM) to stay
// under 160 KB so two workgroups fit per 320 KB WGP.
#define XS_OFF 0
#define QS_OFF (CHK * XS_STRIDE)                 // 50176
#define KS_OFF (QS_OFF + CHK * QK_STRIDE)        // 59392
#define PS_OFF QS_OFF                            // aliases Qs+Ks (17408 <= 18432)
#define VS_OFF (KS_OFF + CHK * QK_STRIDE)        // 68608
#define SMEM_HALVES (VS_OFF + HD * PV_STRIDE)    // 77312 halves = 154624 B

__global__ __launch_bounds__(256, 1)
void qkv_attn_kernel(const float* __restrict__ x, const _Float16* __restrict__ wqkvT,
                     const float* __restrict__ bqkv, _Float16* __restrict__ aout) {
    __shared__ __align__(16) _Float16 smem[SMEM_HALVES];
    _Float16* Xs = smem + XS_OFF;   // x chunk, row=s col=c
    _Float16* Qs = smem + QS_OFF;   // Q row-major (scaled)
    _Float16* Ks = smem + KS_OFF;   // K row-major (== B for Q.K^T)
    _Float16* Ps = smem + PS_OFF;   // softmax probs row-major (aliases Q/K)
    _Float16* Vs = smem + VS_OFF;   // V transposed: Vs[j][s]

    const int b  = blockIdx.y;
    const int ch = blockIdx.x;
    const float* xb = x + (size_t)b * Cdim * Sdim + (size_t)ch * CHK;

    // stage x chunk -> LDS f16  (Xs[s][c] = x[b][c][ch*128+s])
    for (int i = threadIdx.x; i < CHK * Cdim; i += 256) {
        int c = i >> 7, s = i & 127;
        Xs[s * XS_STRIDE + c] = (_Float16)xb[(size_t)c * Sdim + s];
    }
    __syncthreads();

    const int wave = threadIdx.x >> 5;     // 0..7  == M-tile index
    const int lane = threadIdx.x & 31;
    const int l16  = lane & 15;
    const int lhi  = lane >> 4;            // lane half
    const int rowA = wave * 16 + l16;      // A-fragment row
    const int kA   = lhi * 8;              // A-fragment K sub-offset
    const int kB   = lhi * 8;              // B-fragment K sub-offset
    const int rowD = wave * 16 + lhi * 8;  // D rows = rowD..rowD+7

    for (int h = 0; h < NHEAD; ++h) {
        // ---- QKV GEMM: [128 x 384] @ [384 x 192] ----
        v8f acc[12];
        int col[12];
#pragma unroll
        for (int tn = 0; tn < 12; ++tn) {
            int j = tn * 16 + l16;
            col[tn] = (j >> 6) * Cdim + h * HD + (j & 63);   // sel*384 + h*64 + d
            acc[tn] = splat8(bqkv[col[tn]]);
        }
        for (int k0 = 0; k0 < Cdim; k0 += 32) {
            const _Float16* ap = &Xs[rowA * XS_STRIDE + k0 + kA];
            v16h a = load_frag16(ap, ap + 16);
            // groups of 4 N-tiles: 8 parallel b128 loads, then 4 WMMAs
#pragma unroll
            for (int g = 0; g < 3; ++g) {
                v16h bf[4];
#pragma unroll
                for (int t = 0; t < 4; ++t) {
                    const _Float16* bp = wqkvT + (size_t)col[g * 4 + t] * Cdim + k0 + kB;
                    bf[t] = load_frag16(bp, bp + 16);
                }
#pragma unroll
                for (int t = 0; t < 4; ++t)
                    acc[g * 4 + t] = wmma16(a, bf[t], acc[g * 4 + t]);
            }
        }
        // stage Q (x scale), K row-major; V transposed (packed b128 stores)
#pragma unroll
        for (int tn = 0; tn < 4; ++tn) {
            int n = tn * 16 + l16;
#pragma unroll
            for (int r = 0; r < 8; ++r)
                Qs[(rowD + r) * QK_STRIDE + n] = (_Float16)(acc[tn][r] * 0.125f);
        }
#pragma unroll
        for (int tn = 4; tn < 8; ++tn) {
            int n = (tn - 4) * 16 + l16;
#pragma unroll
            for (int r = 0; r < 8; ++r)
                Ks[(rowD + r) * QK_STRIDE + n] = (_Float16)acc[tn][r];
        }
#pragma unroll
        for (int tn = 8; tn < 12; ++tn) {
            int n = (tn - 8) * 16 + l16;
            half8 hv;
#pragma unroll
            for (int r = 0; r < 8; ++r) hv[r] = (_Float16)acc[tn][r];
            *(half8*)&Vs[n * PV_STRIDE + rowD] = hv;   // 8 contiguous s -> one b128
        }
        __syncthreads();

        // ---- logits: Q[128x64] . K^T -> [128x128], wave owns 16 rows ----
        v8f sc[8];
#pragma unroll
        for (int tn = 0; tn < 8; ++tn) sc[tn] = splat8(0.0f);
#pragma unroll
        for (int k0 = 0; k0 < HD; k0 += 32) {
            const _Float16* ap = &Qs[rowA * QK_STRIDE + k0 + kA];
            v16h a = load_frag16(ap, ap + 16);
#pragma unroll
            for (int g = 0; g < 2; ++g) {
                v16h bf[4];
#pragma unroll
                for (int t = 0; t < 4; ++t) {
                    const _Float16* bp = &Ks[((g * 4 + t) * 16 + l16) * QK_STRIDE + k0 + kB];
                    bf[t] = load_frag16(bp, bp + 16);
                }
#pragma unroll
                for (int t = 0; t < 4; ++t)
                    sc[g * 4 + t] = wmma16(a, bf[t], sc[g * 4 + t]);
            }
        }
        __syncthreads();   // Qs/Ks dead everywhere before Ps (alias) is written

        // ---- softmax over 128 cols; each row lives in 16 lanes (fixed vgpr) ----
#pragma unroll
        for (int vr = 0; vr < 8; ++vr) {
            float m = -3.0e38f;
#pragma unroll
            for (int tn = 0; tn < 8; ++tn) m = fmaxf(m, sc[tn][vr]);
#pragma unroll
            for (int msk = 8; msk >= 1; msk >>= 1) m = fmaxf(m, __shfl_xor(m, msk, 32));
            float pv[8], sum = 0.0f;
#pragma unroll
            for (int tn = 0; tn < 8; ++tn) { pv[tn] = __expf(sc[tn][vr] - m); sum += pv[tn]; }
#pragma unroll
            for (int msk = 8; msk >= 1; msk >>= 1) sum += __shfl_xor(sum, msk, 32);
            float rs = 1.0f / sum;
            int row = wave * 16 + lhi * 8 + vr;
#pragma unroll
            for (int tn = 0; tn < 8; ++tn)
                Ps[row * PV_STRIDE + tn * 16 + l16] = (_Float16)(pv[tn] * rs);
        }
        __syncthreads();

        // ---- out = P[128x128] @ V[128x64] ----
        v8f oc[4];
#pragma unroll
        for (int tn = 0; tn < 4; ++tn) oc[tn] = splat8(0.0f);
#pragma unroll
        for (int k0 = 0; k0 < CHK; k0 += 32) {
            const _Float16* ap = &Ps[rowA * PV_STRIDE + k0 + kA];
            v16h a = load_frag16(ap, ap + 16);
            v16h bf[4];
#pragma unroll
            for (int tn = 0; tn < 4; ++tn) {
                const _Float16* bp = &Vs[(tn * 16 + l16) * PV_STRIDE + k0 + kB];
                bf[tn] = load_frag16(bp, bp + 16);
            }
#pragma unroll
            for (int tn = 0; tn < 4; ++tn)
                oc[tn] = wmma16(a, bf[tn], oc[tn]);
        }
        // store head output f16 -> aout[b][s][c], c = h*64 + j
        size_t sbase = (size_t)b * Sdim + (size_t)ch * CHK + rowD;
#pragma unroll
        for (int tn = 0; tn < 4; ++tn) {
            int j = h * HD + tn * 16 + l16;
#pragma unroll
            for (int r = 0; r < 8; ++r)
                aout[(sbase + r) * Cdim + j] = (_Float16)oc[tn][r];
        }
        __syncthreads();   // before next head overwrites Qs/Ks/Vs/Ps
    }
}

// ---------------- kernel 2: projection GEMM, output transposed to (B,C,S) ----------------
__global__ __launch_bounds__(256, 1)
void proj_kernel(const _Float16* __restrict__ aout, const _Float16* __restrict__ wprojT,
                 const float* __restrict__ bproj, float* __restrict__ out) {
    const int b    = blockIdx.z;
    const int sblk = blockIdx.x;   // 512 blocks of 128 rows
    const int nblk = blockIdx.y;   // 6 blocks of 64 cols
    const int wave = threadIdx.x >> 5;
    const int lane = threadIdx.x & 31;
    const int l16  = lane & 15;
    const int lhi  = lane >> 4;

    const _Float16* arow = aout + ((size_t)b * Sdim + (size_t)sblk * CHK) * Cdim;

    v8f acc[4];
    int cout[4];
#pragma unroll
    for (int tn = 0; tn < 4; ++tn) {
        cout[tn] = nblk * 64 + tn * 16 + l16;
        acc[tn] = splat8(bproj[cout[tn]]);
    }
    for (int k0 = 0; k0 < Cdim; k0 += 32) {
        const _Float16* ap = arow + (size_t)(wave * 16 + l16) * Cdim + k0 + lhi * 8;
        v16h a = load_frag16(ap, ap + 16);
        v16h bf[4];
#pragma unroll
        for (int tn = 0; tn < 4; ++tn) {
            const _Float16* bp = wprojT + (size_t)cout[tn] * Cdim + k0 + lhi * 8;
            bf[tn] = load_frag16(bp, bp + 16);
        }
#pragma unroll
        for (int tn = 0; tn < 4; ++tn)
            acc[tn] = wmma16(a, bf[tn], acc[tn]);
    }
    // lane's 8 accumulator rows are 8 consecutive s at fixed cout -> two float4 stores
    size_t srow = (size_t)sblk * CHK + wave * 16 + lhi * 8;
#pragma unroll
    for (int tn = 0; tn < 4; ++tn) {
        size_t obase = ((size_t)b * Cdim + cout[tn]) * Sdim + srow;
        float4 o0 = make_float4(acc[tn][0], acc[tn][1], acc[tn][2], acc[tn][3]);
        float4 o1 = make_float4(acc[tn][4], acc[tn][5], acc[tn][6], acc[tn][7]);
        *(float4*)(out + obase)     = o0;
        *(float4*)(out + obase + 4) = o1;
    }
}

extern "C" void kernel_launch(void* const* d_in, const int* in_sizes, int n_in,
                              void* d_out, int out_size, void* d_ws, size_t ws_size,
                              hipStream_t stream) {
    const float* x     = (const float*)d_in[0];
    const float* wqkv  = (const float*)d_in[1];
    const float* bqkv  = (const float*)d_in[2];
    const float* wproj = (const float*)d_in[3];
    const float* bproj = (const float*)d_in[4];
    float* out = (float*)d_out;

    _Float16* ws      = (_Float16*)d_ws;
    _Float16* wqkvT   = ws;                                   // 1152*384 halves
    _Float16* wprojT  = wqkvT + (size_t)C3 * Cdim;            // 384*384 halves
    _Float16* aout    = wprojT + (size_t)Cdim * Cdim;         // 2*65536*384 halves

    prep_weights<<<dim3(512), 256, 0, stream>>>(wqkv, wproj, wqkvT, wprojT);
    qkv_attn_kernel<<<dim3(NCHK, Bsz), 256, 0, stream>>>(x, wqkvT, bqkv, aout);
    proj_kernel<<<dim3(NCHK, NHEAD, Bsz), 256, 0, stream>>>(aout, wprojT, bproj, out);
}
